// ODEFullBlock_71330816852799
// MI455X (gfx1250) — matrix operand order repaired and checked by reading, not compiled, not run on hardware
//
#include <hip/hip_runtime.h>

typedef _Float16 half_t;
typedef _Float16 v16h __attribute__((ext_vector_type(16)));
typedef _Float16 v8h  __attribute__((ext_vector_type(8)));
typedef float    v8f  __attribute__((ext_vector_type(8)));
typedef int      v4i  __attribute__((ext_vector_type(4)));

#define GN_EPS 1e-5f
#define B_SZ   256

// ---------------------------------------------------------------------------
// CDNA5 async global->LDS copy (ASYNCcnt-tracked DMA, bypasses VGPRs).
// Confirmed signature: (int4 AS1* src, int4 AS3* dst, imm offset, imm cpol).
// Guarded: falls back to a VGPR-mediated 16B copy if the builtin is absent.
// ---------------------------------------------------------------------------
__device__ __forceinline__ void async_cp16(half_t* dst, const half_t* src) {
#if __has_builtin(__builtin_amdgcn_global_load_async_to_lds_b128)
  __builtin_amdgcn_global_load_async_to_lds_b128(
      (__attribute__((address_space(1))) v4i*)(unsigned long long)src,
      (__attribute__((address_space(3))) v4i*)(v4i*)dst, 0, 0);
#else
  *(v8h*)dst = *(const v8h*)src;
#endif
}

#if __has_builtin(__builtin_amdgcn_s_wait_asynccnt)
  #define WAIT_ASYNC() __builtin_amdgcn_s_wait_asynccnt(0)
#else
  #define WAIT_ASYNC() asm volatile("s_wait_asynccnt 0" ::: "memory")
#endif

// ---------------------------------------------------------------------------
// Weight packing into WMMA B-fragment order.
// v_wmma_f32_16x16x32_f16 B (32x16, KxN), wave32:
//   lane l: column n = l&15 ; element j (0..15): K = ((l>>4)&1)*8 + (j&7) + (j>>3)*16
// packed layout: wp[tap][kc][wave][lane][j]  (contiguous v16h per lane)
// ---------------------------------------------------------------------------
__global__ void pack_w44_k(const float* __restrict__ w, half_t* __restrict__ out) {
  int idx = blockIdx.x * blockDim.x + threadIdx.x;
  if (idx >= 16 * 2 * 4 * 32 * 16) return;
  int j  = idx & 15;
  int r  = idx >> 4;
  int l  = r & 31; r >>= 5;
  int wv = r & 3;  r >>= 2;
  int kc = r & 1;
  int tap = r >> 1;                       // 0..15  (4x4 kernel)
  int o = wv * 16 + (l & 15);
  int i = kc * 32 + ((l >> 4) & 1) * 8 + (j & 7) + (j >> 3) * 16;
  out[idx] = (half_t)w[((o * 64 + i) * 4 + (tap >> 2)) * 4 + (tap & 3)];
}

// 3x3 conv weights with 65 input channels (channel 0 = time).
// Packs channels 1..64 as B fragments; channel 0 folded into a per-border-class
// bias table wtsum[9][64] (sum of w_t over taps valid at that pixel class).
__global__ void pack_w33_k(const float* __restrict__ w, half_t* __restrict__ out,
                           float* __restrict__ wtsum) {
  int idx = blockIdx.x * blockDim.x + threadIdx.x;
  const int total = 9 * 2 * 4 * 32 * 16;
  if (idx < total) {
    int j  = idx & 15;
    int r  = idx >> 4;
    int l  = r & 31; r >>= 5;
    int wv = r & 3;  r >>= 2;
    int kc = r & 1;
    int tap = r >> 1;                     // 0..8 (3x3 kernel)
    int o = wv * 16 + (l & 15);
    int i = kc * 32 + ((l >> 4) & 1) * 8 + (j & 7) + (j >> 3) * 16;       // 0..63 -> input ch i+1
    out[idx] = (half_t)w[((o * 65 + (i + 1)) * 3 + (tap / 3)) * 3 + (tap % 3)];
  } else if (idx < total + 9 * 64) {
    int r = idx - total;
    int cls = r >> 6, o = r & 63;
    int cy = cls / 3, cx = cls % 3;       // 0: edge-lo, 1: interior, 2: edge-hi
    float s = 0.f;
    for (int ty = 0; ty < 3; ++ty)
      for (int tx = 0; tx < 3; ++tx) {
        bool vy = !((cy == 0 && ty == 0) || (cy == 2 && ty == 2));
        bool vx = !((cx == 0 && tx == 0) || (cx == 2 && tx == 2));
        if (vy && vx) s += w[((o * 65) * 3 + ty) * 3 + tx];
      }
    wtsum[r] = s;
  }
}

// Zeroed 16B guard rows: out-of-image taps DMA from here (uniform EXEC, no
// divergent zero-fill of LDS needed).
__global__ void zero_pad_k(half_t* __restrict__ z) {
  z[threadIdx.x] = (half_t)0.f;    // 128 halves = 256 B
}

// ---------------------------------------------------------------------------
// conv1: 3->64, 3x3, stride 1, no pad.  x is NCHW fp32, out NHWC fp32.
// K=27 is too small for WMMA; direct VALU (0.8 GFLOP, negligible).
// ---------------------------------------------------------------------------
__global__ void conv1_k(const float* __restrict__ x, const float* __restrict__ w,
                        const float* __restrict__ bias, float* __restrict__ out) {
  int idx = blockIdx.x * blockDim.x + threadIdx.x;   // B*30*30*64
  int o  = idx & 63;
  int t  = idx >> 6;
  int px = t % 30; int t2 = t / 30;
  int py = t2 % 30; int b = t2 / 30;
  float acc = bias[o];
  for (int ci = 0; ci < 3; ++ci)
    for (int ky = 0; ky < 3; ++ky)
      for (int kx = 0; kx < 3; ++kx)
        acc += x[((b * 3 + ci) * 32 + py + ky) * 32 + px + kx] *
               w[((o * 3 + ci) * 3 + ky) * 3 + kx];
  out[idx] = acc;
}

// ---------------------------------------------------------------------------
// GroupNorm (32 groups over 64 ch -> 2 ch/group), NHWC fp32 in.
// f32-out variant (optional ReLU) feeds GN-stat consumers / RK math;
// f16-out variant (always ReLU) feeds the WMMA convs so their staging can be
// a pure byte-copy async DMA.  One block per (batch, group); in-place safe.
// ---------------------------------------------------------------------------
__device__ __forceinline__ void gn_stats(const float* in, int b, int c0, int HW,
                                         int n, float* s1, float* s2,
                                         float& mean, float& inv) {
  int tid = threadIdx.x;
  float sum = 0.f, ssq = 0.f;
  for (int i = tid; i < n; i += 128) {
    int p = i >> 1, c = c0 + (i & 1);
    float v = in[(size_t)(b * HW + p) * 64 + c];
    sum += v; ssq += v * v;
  }
  s1[tid] = sum; s2[tid] = ssq;
  __syncthreads();
  for (int off = 64; off > 0; off >>= 1) {
    if (tid < off) { s1[tid] += s1[tid + off]; s2[tid] += s2[tid + off]; }
    __syncthreads();
  }
  mean = s1[0] / (float)n;
  float var = s2[0] / (float)n - mean * mean;
  inv = rsqrtf(var + GN_EPS);
}

__global__ void groupnorm_k(const float* in, float* out,
                            const float* __restrict__ gw, const float* __restrict__ gb,
                            int HW, int relu) {
  __shared__ float s1[128], s2[128];
  int b = blockIdx.x >> 5, g = blockIdx.x & 31, c0 = g * 2, n = HW * 2;
  float mean, inv;
  gn_stats(in, b, c0, HW, n, s1, s2, mean, inv);
  for (int i = threadIdx.x; i < n; i += 128) {
    int p = i >> 1, c = c0 + (i & 1);
    size_t id = (size_t)(b * HW + p) * 64 + c;
    float v = (in[id] - mean) * inv * gw[c] + gb[c];
    if (relu) v = fmaxf(v, 0.f);
    out[id] = v;
  }
}

__global__ void groupnorm_h_k(const float* in, half_t* out,
                              const float* __restrict__ gw, const float* __restrict__ gb,
                              int HW) {
  __shared__ float s1[128], s2[128];
  int b = blockIdx.x >> 5, g = blockIdx.x & 31, c0 = g * 2, n = HW * 2;
  float mean, inv;
  gn_stats(in, b, c0, HW, n, s1, s2, mean, inv);
  for (int i = threadIdx.x; i < n; i += 128) {
    int p = i >> 1, c = c0 + (i & 1);
    size_t id = (size_t)(b * HW + p) * 64 + c;
    float v = (in[id] - mean) * inv * gw[c] + gb[c];
    out[id] = (half_t)fmaxf(v, 0.f);
  }
}

// ---------------------------------------------------------------------------
// WMMA implicit-GEMM conv, 4x4 kernel, stride 2, pad 1, 64->64.
// Input NHWC f16 (post-GN+ReLU), output NHWC fp32.
// Block = 128 threads (4 wave32), computes 16 pixels x 64 out-channels.
// A patches staged to LDS by GLOBAL_LOAD_ASYNC_TO_LDS_B128 (ASYNCcnt DMA);
// out-of-image taps are DMA'd from a zeroed guard row -> uniform EXEC.
// A fragment (16x32 f16, ISA 7.12.2): lane<16: row=lane, K {0-7,16-23};
// lane>=16: row=lane-16, K {8-15,24-31}.
// ---------------------------------------------------------------------------
__global__ __launch_bounds__(128)
void conv44s2_wmma_k(const half_t* __restrict__ in, float* __restrict__ out,
                     const half_t* __restrict__ wp, const float* __restrict__ bias,
                     const half_t* __restrict__ zpad,
                     int Hi, int Wi, int Ho, int Wo) {
  __shared__ half_t a_lds[16][16][64];            // 32 KB
  int tid = threadIdx.x;
  int tileM = blockIdx.x;
  int HWo = Ho * Wo;
  __builtin_prefetch(wp, 0, 0);

  // 16 px * 16 taps * 8 chunks of 16B = 2048 lane-ops (uniform: 2048 % 128 == 0)
  for (int s = tid; s < 16 * 16 * 8; s += 128) {
    int ck = s & 7;
    int r  = s >> 3;
    int tap = r & 15, px = r >> 4;
    int m = tileM * 16 + px;
    int b = m / HWo; int rem = m % HWo;
    int oy = rem / Wo, ox = rem % Wo;
    int iy = oy * 2 - 1 + (tap >> 2);
    int ix = ox * 2 - 1 + (tap & 3);
    bool valid = (iy >= 0 && iy < Hi && ix >= 0 && ix < Wi);
    const half_t* src = valid
        ? in + ((size_t)(b * Hi + iy) * Wi + ix) * 64 + ck * 8
        : zpad + ck * 8;
    async_cp16(&a_lds[px][tap][ck * 8], src);
  }
  WAIT_ASYNC();
  __syncthreads();

  int wv = tid >> 5, lane = tid & 31;
  int mrow = lane & 15, hiK = (lane >> 4) & 1;
  v8f c;
  for (int j = 0; j < 8; ++j) c[j] = 0.f;

  for (int tap = 0; tap < 16; ++tap) {
    for (int kc = 0; kc < 2; ++kc) {
      const half_t* ar = &a_lds[mrow][tap][kc * 32 + hiK * 8];
      v8h a_lo = *(const v8h*)ar;
      v8h a_hi = *(const v8h*)(ar + 16);
      v16h a;
      for (int j = 0; j < 8; ++j) { a[j] = a_lo[j]; a[j + 8] = a_hi[j]; }
      const half_t* bp = wp + (size_t)((((tap * 2 + kc) * 4 + wv) * 32) + lane) * 16;
      v16h bm = *(const v16h*)bp;
      c = __builtin_amdgcn_wmma_f32_16x16x32_f16(false, a, false, bm, (short)0, c,
                                                 false, false);
    }
  }

  int ncol = wv * 16 + (lane & 15);
  float bv = bias[ncol];
  for (int j = 0; j < 8; ++j) {
    size_t m = (size_t)tileM * 16 + j + hiK * 8;   // C/D: M = vgpr + 8*(lane>=16)
    out[m * 64 + ncol] = c[j] + bv;
  }
}

// ---------------------------------------------------------------------------
// WMMA implicit-GEMM ODE conv: 3x3, stride 1, pad 1, 7x7 spatial, 64(+t)->64.
// Input NHWC f16, output NHWC fp32.  Time channel handled analytically in the
// epilogue via wtsum[border_class].
// ---------------------------------------------------------------------------
__global__ __launch_bounds__(128)
void conv33_ode_wmma_k(const half_t* __restrict__ in, float* __restrict__ out,
                       const half_t* __restrict__ wp, const float* __restrict__ bias,
                       const float* __restrict__ wtsum, const half_t* __restrict__ zpad,
                       float t) {
  __shared__ half_t a_lds[16][9][64];             // 18 KB
  const int Hi = 7, Wi = 7, HWo = 49;
  int tid = threadIdx.x;
  int tileM = blockIdx.x;
  __builtin_prefetch(wp, 0, 0);

  // 16 px * 9 taps * 8 chunks = 1152 lane-ops (1152 % 128 == 0)
  for (int s = tid; s < 16 * 9 * 8; s += 128) {
    int ck = s & 7;
    int r  = s >> 3;
    int tap = r % 9, px = r / 9;
    int m = tileM * 16 + px;
    int b = m / HWo; int rem = m % HWo;
    int oy = rem / 7, ox = rem % 7;
    int iy = oy - 1 + (tap / 3);
    int ix = ox - 1 + (tap % 3);
    bool valid = (iy >= 0 && iy < Hi && ix >= 0 && ix < Wi);
    const half_t* src = valid
        ? in + ((size_t)(b * Hi + iy) * Wi + ix) * 64 + ck * 8
        : zpad + ck * 8;
    async_cp16(&a_lds[px][tap][ck * 8], src);
  }
  WAIT_ASYNC();
  __syncthreads();

  int wv = tid >> 5, lane = tid & 31;
  int mrow = lane & 15, hiK = (lane >> 4) & 1;
  v8f c;
  for (int j = 0; j < 8; ++j) c[j] = 0.f;

  for (int tap = 0; tap < 9; ++tap) {
    for (int kc = 0; kc < 2; ++kc) {
      const half_t* ar = &a_lds[mrow][tap][kc * 32 + hiK * 8];
      v8h a_lo = *(const v8h*)ar;
      v8h a_hi = *(const v8h*)(ar + 16);
      v16h a;
      for (int j = 0; j < 8; ++j) { a[j] = a_lo[j]; a[j + 8] = a_hi[j]; }
      const half_t* bp = wp + (size_t)((((tap * 2 + kc) * 4 + wv) * 32) + lane) * 16;
      v16h bm = *(const v16h*)bp;
      c = __builtin_amdgcn_wmma_f32_16x16x32_f16(false, a, false, bm, (short)0, c,
                                                 false, false);
    }
  }

  int ncol = wv * 16 + (lane & 15);
  float bv = bias[ncol];
  for (int j = 0; j < 8; ++j) {
    size_t m = (size_t)tileM * 16 + j + hiK * 8;
    int rem = (int)(m % 49);
    int y = rem / 7, x = rem % 7;
    int cls = ((y == 0) ? 0 : ((y == 6) ? 2 : 1)) * 3 +
              ((x == 0) ? 0 : ((x == 6) ? 2 : 1));
    out[m * 64 + ncol] = c[j] + bv + t * wtsum[cls * 64 + ncol];
  }
}

// ---------------------------------------------------------------------------
// out = y + h * (c1*p1 + ... + c6*p6)   (RK stage combination / final update)
// ---------------------------------------------------------------------------
__global__ void axpy6_k(float* __restrict__ out, const float* __restrict__ y,
                        const float* __restrict__ p1, const float* __restrict__ p2,
                        const float* __restrict__ p3, const float* __restrict__ p4,
                        const float* __restrict__ p5, const float* __restrict__ p6,
                        float c1, float c2, float c3, float c4, float c5, float c6,
                        float h, int n) {
  int i = blockIdx.x * blockDim.x + threadIdx.x;
  if (i >= n) return;
  float s = c1 * p1[i] + c2 * p2[i] + c3 * p3[i] + c4 * p4[i] + c5 * p5[i] + c6 * p6[i];
  out[i] = y[i] + h * s;
}

// ---------------------------------------------------------------------------
// Global average pool over 49 pixels + linear 64->10. One block per batch elem.
// ---------------------------------------------------------------------------
__global__ void pool_linear_k(const float* __restrict__ in, const float* __restrict__ lw,
                              const float* __restrict__ lb, float* __restrict__ out) {
  __shared__ float ms[64];
  int b = blockIdx.x;
  int c = threadIdx.x;      // 64 threads
  float s = 0.f;
  for (int p = 0; p < 49; ++p) s += in[((size_t)b * 49 + p) * 64 + c];
  ms[c] = s * (1.0f / 49.0f);
  __syncthreads();
  if (c < 10) {
    float acc = lb[c];
    for (int k = 0; k < 64; ++k) acc += ms[k] * lw[c * 64 + k];
    out[b * 10 + c] = acc;
  }
}

// ---------------------------------------------------------------------------
extern "C" void kernel_launch(void* const* d_in, const int* in_sizes, int n_in,
                              void* d_out, int out_size, void* d_ws, size_t ws_size,
                              hipStream_t stream) {
  (void)in_sizes; (void)n_in; (void)out_size; (void)ws_size;

  const float* x      = (const float*)d_in[0];
  const float* c1_w   = (const float*)d_in[1];
  const float* c1_b   = (const float*)d_in[2];
  const float* g1_w   = (const float*)d_in[3];
  const float* g1_b   = (const float*)d_in[4];
  const float* c2_w   = (const float*)d_in[5];
  const float* c2_b   = (const float*)d_in[6];
  const float* g2_w   = (const float*)d_in[7];
  const float* g2_b   = (const float*)d_in[8];
  const float* c3_w   = (const float*)d_in[9];
  const float* c3_b   = (const float*)d_in[10];
  const float* oa_w   = (const float*)d_in[11];
  const float* oa_b   = (const float*)d_in[12];
  const float* occ1_w = (const float*)d_in[13];
  const float* occ1_b = (const float*)d_in[14];
  const float* ob_w   = (const float*)d_in[15];
  const float* ob_b   = (const float*)d_in[16];
  const float* occ2_w = (const float*)d_in[17];
  const float* occ2_b = (const float*)d_in[18];
  const float* oc_w   = (const float*)d_in[19];
  const float* oc_b   = (const float*)d_in[20];
  const float* g3_w   = (const float*)d_in[21];
  const float* g3_b   = (const float*)d_in[22];
  const float* lin_w  = (const float*)d_in[23];
  const float* lin_b  = (const float*)d_in[24];

  // ---- workspace carve (bump allocator, 256B aligned) ----
  char*  ws  = (char*)d_ws;
  size_t off = 0;
  auto alloc = [&](size_t bytes) -> void* {
    void* p = ws + off;
    off += (bytes + 255) & ~(size_t)255;
    return p;
  };
  const int NE = B_SZ * 49 * 64;                        // ODE state elements
  float*  y     = (float*)alloc((size_t)NE * 4);
  float*  ytmp  = (float*)alloc((size_t)NE * 4);
  float*  k1    = (float*)alloc((size_t)NE * 4);
  float*  k2    = (float*)alloc((size_t)NE * 4);
  float*  k3    = (float*)alloc((size_t)NE * 4);
  float*  k4    = (float*)alloc((size_t)NE * 4);
  float*  k5    = (float*)alloc((size_t)NE * 4);
  float*  k6    = (float*)alloc((size_t)NE * 4);
  float*  tB    = (float*)alloc((size_t)NE * 4);        // conv outputs / GN3 out
  half_t* tAh   = (half_t*)alloc((size_t)NE * 2);       // GN(f16) -> ODE convs
  float*  act1  = (float*)alloc((size_t)B_SZ * 900 * 64 * 4);  // conv1 out fp32
  half_t* act1h = (half_t*)alloc((size_t)B_SZ * 900 * 64 * 2); // GN1 out f16
  float*  act2  = (float*)alloc((size_t)B_SZ * 225 * 64 * 4);  // conv2 out fp32
  half_t* act2h = (half_t*)alloc((size_t)B_SZ * 225 * 64 * 2); // GN2 out f16
  half_t* c2p   = (half_t*)alloc((size_t)16 * 2 * 4 * 32 * 16 * 2);
  half_t* c3p   = (half_t*)alloc((size_t)16 * 2 * 4 * 32 * 16 * 2);
  half_t* o1p   = (half_t*)alloc((size_t)9 * 2 * 4 * 32 * 16 * 2);
  half_t* o2p   = (half_t*)alloc((size_t)9 * 2 * 4 * 32 * 16 * 2);
  float*  wts1  = (float*)alloc((size_t)9 * 64 * 4);
  float*  wts2  = (float*)alloc((size_t)9 * 64 * 4);
  half_t* zpad  = (half_t*)alloc((size_t)128 * 2);      // zeroed guard rows

  const int GN_GRID = B_SZ * 32;

  // ---- weight packing + guard init ----
  zero_pad_k<<<1, 128, 0, stream>>>(zpad);
  pack_w44_k<<<(16 * 2 * 4 * 32 * 16 + 255) / 256, 256, 0, stream>>>(c2_w, c2p);
  pack_w44_k<<<(16 * 2 * 4 * 32 * 16 + 255) / 256, 256, 0, stream>>>(c3_w, c3p);
  pack_w33_k<<<(9 * 2 * 4 * 32 * 16 + 9 * 64 + 255) / 256, 256, 0, stream>>>(occ1_w, o1p, wts1);
  pack_w33_k<<<(9 * 2 * 4 * 32 * 16 + 9 * 64 + 255) / 256, 256, 0, stream>>>(occ2_w, o2p, wts2);

  // ---- feature extractor ----
  conv1_k<<<(B_SZ * 900 * 64) / 256, 256, 0, stream>>>(x, c1_w, c1_b, act1);
  groupnorm_h_k<<<GN_GRID, 128, 0, stream>>>(act1, act1h, g1_w, g1_b, 900);
  conv44s2_wmma_k<<<(B_SZ * 225) / 16, 128, 0, stream>>>(act1h, act2, c2p, c2_b, zpad,
                                                         30, 30, 15, 15);
  groupnorm_h_k<<<GN_GRID, 128, 0, stream>>>(act2, act2h, g2_w, g2_b, 225);
  conv44s2_wmma_k<<<(B_SZ * 49) / 16, 128, 0, stream>>>(act2h, y, c3p, c3_b, zpad,
                                                        15, 15, 7, 7);

  // ---- ODE solve: 6 fixed dopri5 steps on t in [0,1] ----
  const int ODE_GRID = (B_SZ * 49) / 16;        // 784 M-tiles
  const int AX_GRID  = (NE + 255) / 256;
  const float hs = 1.0f / 6.0f;

  auto feval = [&](float t, const float* yin, float* kout) {
    groupnorm_h_k<<<GN_GRID, 128, 0, stream>>>(yin, tAh, oa_w, oa_b, 49);
    conv33_ode_wmma_k<<<ODE_GRID, 128, 0, stream>>>(tAh, tB, o1p, occ1_b, wts1, zpad, t);
    groupnorm_h_k<<<GN_GRID, 128, 0, stream>>>(tB, tAh, ob_w, ob_b, 49);
    conv33_ode_wmma_k<<<ODE_GRID, 128, 0, stream>>>(tAh, tB, o2p, occ2_b, wts2, zpad, t);
    groupnorm_k<<<GN_GRID, 128, 0, stream>>>(tB, kout, oc_w, oc_b, 49, 0);
  };

  for (int i = 0; i < 6; ++i) {
    float t0 = i * hs;
    feval(t0, y, k1);
    axpy6_k<<<AX_GRID, 256, 0, stream>>>(ytmp, y, k1, k1, k1, k1, k1, k1,
        0.2f, 0.f, 0.f, 0.f, 0.f, 0.f, hs, NE);
    feval(t0 + hs * 0.2f, ytmp, k2);
    axpy6_k<<<AX_GRID, 256, 0, stream>>>(ytmp, y, k1, k2, k1, k1, k1, k1,
        (float)(3.0 / 40.0), (float)(9.0 / 40.0), 0.f, 0.f, 0.f, 0.f, hs, NE);
    feval(t0 + hs * 0.3f, ytmp, k3);
    axpy6_k<<<AX_GRID, 256, 0, stream>>>(ytmp, y, k1, k2, k3, k1, k1, k1,
        (float)(44.0 / 45.0), (float)(-56.0 / 15.0), (float)(32.0 / 9.0),
        0.f, 0.f, 0.f, hs, NE);
    feval(t0 + hs * 0.8f, ytmp, k4);
    axpy6_k<<<AX_GRID, 256, 0, stream>>>(ytmp, y, k1, k2, k3, k4, k1, k1,
        (float)(19372.0 / 6561.0), (float)(-25360.0 / 2187.0),
        (float)(64448.0 / 6561.0), (float)(-212.0 / 729.0), 0.f, 0.f, hs, NE);
    feval(t0 + hs * (8.0f / 9.0f), ytmp, k5);
    axpy6_k<<<AX_GRID, 256, 0, stream>>>(ytmp, y, k1, k2, k3, k4, k5, k1,
        (float)(9017.0 / 3168.0), (float)(-355.0 / 33.0),
        (float)(46732.0 / 5247.0), (float)(49.0 / 176.0),
        (float)(-5103.0 / 18656.0), 0.f, hs, NE);
    feval(t0 + hs, ytmp, k6);
    axpy6_k<<<AX_GRID, 256, 0, stream>>>(y, y, k1, k2, k3, k4, k5, k6,
        (float)(35.0 / 384.0), 0.f, (float)(500.0 / 1113.0),
        (float)(125.0 / 192.0), (float)(-2187.0 / 6784.0),
        (float)(11.0 / 84.0), hs, NE);
  }

  // ---- head: GN + ReLU, avg-pool(7x7), linear 64->10 ----
  groupnorm_k<<<GN_GRID, 128, 0, stream>>>(y, tB, g3_w, g3_b, 49, 1);
  pool_linear_k<<<B_SZ, 64, 0, stream>>>(tB, lin_w, lin_b, (float*)d_out);
}
